// NVTXDenoiseStepWrapper_50397146251510
// MI455X (gfx1250) — compile-verified, matrix operand order checked by reading, not run on hardware
//
#include <hip/hip_runtime.h>
#include <math.h>

// ---- model constants ----
#define BATCH 8
#define HID   1024
#define NH    8
#define KVH   1
#define HD    256
#define NLAY  18
#define MLP   4096
#define AH    50
#define AD    32
#define PFX   1037
#define TOTK  (PFX + AH)        // 1087
#define NTOK  (BATCH * AH)      // 400
#define ATT_SCALE 0.0625f       // 256^-0.5
#define NEGV  -2.3819763e38f
#define TWO_PI 6.28318530717958647692f

typedef __attribute__((ext_vector_type(16))) __bf16 v16bf;
typedef __attribute__((ext_vector_type(8)))  float  v8f;
typedef __attribute__((ext_vector_type(4)))  float  v4fv;
typedef __attribute__((ext_vector_type(4)))  __bf16 v4bfv;
typedef __attribute__((ext_vector_type(2)))  float  v2fv;
typedef __attribute__((ext_vector_type(2)))  __bf16 v2bf;

// packed fp32x2 -> bf16x2 (native v_cvt_pk_bf16_f32, RNE)
__device__ __forceinline__ v2bf pack2(float a, float b) {
  v2fv t = {a, b};
  return __builtin_convertvector(t, v2bf);
}

// Load a 16-bit A/B WMMA fragment from a contiguous fp32 row (32B-aligned).
// Element e holds K = (e&7) + 8*g + 16*(e>>3); caller passes off = kbase + 8*g.
__device__ __forceinline__ v16bf load_frag_f32(const float* p, int off) {
  v16bf f;
  if (p) {
    const v4fv* q0 = (const v4fv*)(p + off);
    const v4fv* q1 = (const v4fv*)(p + off + 16);
    v4bfv c0 = __builtin_convertvector(q0[0], v4bfv);
    v4bfv c1 = __builtin_convertvector(q0[1], v4bfv);
    v4bfv c2 = __builtin_convertvector(q1[0], v4bfv);
    v4bfv c3 = __builtin_convertvector(q1[1], v4bfv);
#pragma unroll
    for (int e = 0; e < 4; ++e) {
      f[e] = c0[e]; f[4 + e] = c1[e]; f[8 + e] = c2[e]; f[12 + e] = c3[e];
    }
  } else {
#pragma unroll
    for (int e = 0; e < 16; ++e) f[e] = (__bf16)0.0f;
  }
  return f;
}

// Load a 16-bit fragment from an LDS row (contiguous along K).
__device__ __forceinline__ v16bf lds_frag(const __bf16* row, int off) {
  v16bf f;
#pragma unroll
  for (int e = 0; e < 8; ++e) { f[e] = row[off + e]; f[e + 8] = row[off + 16 + e]; }
  return f;
}

// =====================================================================
// GEMM: C[z][m][n] = A[m][:] @ W[z][:][n]; fp32 in/out, bf16 WMMA math.
// K, N compile-time -> constant-offset addressing (no u64 muls).
// grid = (ceil(Mreal/32), N/128, Z); block = 128 (4 waves).
// Each wave: 32(M) x 32(N) via 4 wmma per k-step.
// =====================================================================
template <int K, int N>
__global__ void gemm_kernel(const float* __restrict__ Aq, const float* __restrict__ Wq_,
                            float* __restrict__ Cq, int Mreal,
                            long long wstride, long long cstride)
{
  __shared__ __bf16 WT[128][40];   // [n][k] transposed tile; 80B row stride
  const int tid  = threadIdx.x;
  const int wave = tid >> 5, lane = tid & 31;
  const int r = lane & 15, g = lane >> 4;
  const int m0 = blockIdx.x * 32;
  const int n0 = blockIdx.y * 128;
  const float* Wz = Wq_ + (long long)blockIdx.z * wstride;
  float*       Cz = Cq  + (long long)blockIdx.z * cstride;

  v8f acc00 = {0,0,0,0,0,0,0,0};
  v8f acc01 = {0,0,0,0,0,0,0,0};
  v8f acc10 = {0,0,0,0,0,0,0,0};
  v8f acc11 = {0,0,0,0,0,0,0,0};
  const bool a0ok = (m0 + r) < Mreal;
  const bool a1ok = (m0 + 16 + r) < Mreal;
  const float* ar0 = Aq + (long long)(m0 + r) * K;
  const float* ar1 = Aq + (long long)(m0 + 16 + r) * K;

  for (int k0 = 0; k0 < K; k0 += 32) {
    // cooperative load W[k0:k0+32][n0:n0+128] -> WT[n][k]; nn = tid, constant strides
    {
      const float* wp = Wz + (long long)k0 * N + n0 + tid;
      v2bf* lrow = (v2bf*)&WT[tid][0];
#pragma unroll
      for (int it = 0; it < 16; ++it)
        lrow[it] = pack2(wp[(long long)(2 * it) * N], wp[(long long)(2 * it + 1) * N]);
    }
    if (k0 + 32 < K)   // hint next tile into cache (global_prefetch_b8)
      __builtin_prefetch(&Wz[(long long)(k0 + 32) * N + n0 + tid], 0, 1);
    __syncthreads();

    v16bf af0 = load_frag_f32(a0ok ? (ar0 + k0) : nullptr, 8 * g);
    v16bf af1 = load_frag_f32(a1ok ? (ar1 + k0) : nullptr, 8 * g);

    v16bf b0 = lds_frag(&WT[wave * 32 + r][0],      8 * g);
    v16bf b1 = lds_frag(&WT[wave * 32 + 16 + r][0], 8 * g);

    acc00 = __builtin_amdgcn_wmma_f32_16x16x32_bf16(false, af0, false, b0, (short)0, acc00, false, false);
    acc01 = __builtin_amdgcn_wmma_f32_16x16x32_bf16(false, af0, false, b1, (short)0, acc01, false, false);
    acc10 = __builtin_amdgcn_wmma_f32_16x16x32_bf16(false, af1, false, b0, (short)0, acc10, false, false);
    acc11 = __builtin_amdgcn_wmma_f32_16x16x32_bf16(false, af1, false, b1, (short)0, acc11, false, false);
    __syncthreads();
  }

#pragma unroll
  for (int j = 0; j < 8; ++j) {
    int mm0 = m0 + j + 8 * g;      // C layout: row = VGPR_idx + 8*lanegroup
    int mm1 = mm0 + 16;
    if (mm0 < Mreal) {
      Cz[(long long)mm0 * N + n0 + wave * 32 + r]      = acc00[j];
      Cz[(long long)mm0 * N + n0 + wave * 32 + 16 + r] = acc01[j];
    }
    if (mm1 < Mreal) {
      Cz[(long long)mm1 * N + n0 + wave * 32 + r]      = acc10[j];
      Cz[(long long)mm1 * N + n0 + wave * 32 + 16 + r] = acc11[j];
    }
  }
}

// =====================================================================
// Flash attention, one block per (head, batch); wave w = query tile w.
// S^T = K_tile @ Q^T so that P^T (C-layout) feeds the PV wmma
// (O^T = V^T @ P^T) as a B-fragment with no cross-lane movement.
// Q (once), K and V tiles staged in LDS as bf16 -> persistent VGPRs are
// just the 16 O^T accumulators (no spills).
// =====================================================================
__global__ void __launch_bounds__(128, 1)
attn_kernel(const float* __restrict__ qb, const float* __restrict__ kb_,
            const float* __restrict__ vb_, const float* __restrict__ pk,
            const float* __restrict__ pv, const unsigned char* __restrict__ msk,
            float* __restrict__ outb, int layer)
{
  __shared__ __bf16 QT[64][264];   // [query][d]         (528B row stride)
  __shared__ __bf16 KT[32][264];   // [key_in_block][d]  (528B row stride)
  __shared__ __bf16 VT[HD][40];    // V^T: [d][key_in_block] (80B row stride)
  const int tid  = threadIdx.x;
  const int wave = tid >> 5, lane = tid & 31;
  const int r = lane & 15, g = lane >> 4;
  const int h = blockIdx.x;
  const int b = blockIdx.y;
  const int qi = wave * 16 + r;   // this lane-column's query
  const long long pkv = ((long long)(b * NLAY + layer)) * PFX * HD;

  const int dK = 2 * tid;   // this thread's d-pair for Q/K tiles (0..254)

  // ---- cooperative Q tile (once): 64 query rows x 256 d -> QT ----
  for (int it = 0; it < 64; ++it) {
    const float* rp = (it < AH) ? qb + ((long long)(b * AH + it)) * (NH * HD) + h * HD : nullptr;
    v2fv qv = rp ? *(const v2fv*)(rp + dK) : (v2fv){0.f, 0.f};
    *(v2bf*)&QT[it][dK] = pack2(qv[0], qv[1]);
  }
  const __bf16* qp = &QT[wave * 16 + r][0];   // B-frag row for this lane (col = query)

  v8f Ot[16];
#pragma unroll
  for (int i = 0; i < 16; ++i) Ot[i] = (v8f){0,0,0,0,0,0,0,0};
  float mrun = -1e30f, lrun = 0.0f;

  for (int kb0 = 0; kb0 < TOTK; kb0 += 32) {
    // ---- cooperative K tile: [32 keys x 256 d] -> KT[key][d] ----
    for (int it = 0; it < 32; ++it) {
      int j = kb0 + it;
      const float* rp = (j < PFX)  ? pk + pkv + (long long)j * HD
                      : (j < TOTK) ? kb_ + (long long)(b * AH + (j - PFX)) * HD : nullptr;
      v2fv kv = rp ? *(const v2fv*)(rp + dK) : (v2fv){0.f, 0.f};
      *(v2bf*)&KT[it][dK] = pack2(kv[0], kv[1]);
    }
    // ---- cooperative V tile: [32 keys x 256 d] -> VT[d][key] (transposed) ----
    for (int k2 = 0; k2 < 16; ++k2) {
      int j0 = kb0 + 2 * k2, j1 = j0 + 1;
      const float* rp0 = (j0 < PFX)  ? pv + pkv + (long long)j0 * HD
                       : (j0 < TOTK) ? vb_ + (long long)(b * AH + (j0 - PFX)) * HD : nullptr;
      const float* rp1 = (j1 < PFX)  ? pv + pkv + (long long)j1 * HD
                       : (j1 < TOTK) ? vb_ + (long long)(b * AH + (j1 - PFX)) * HD : nullptr;
#pragma unroll
      for (int half = 0; half < 2; ++half) {
        int d = tid + 128 * half;
        float va  = rp0 ? rp0[d] : 0.f;
        float vb2 = rp1 ? rp1[d] : 0.f;
        *(v2bf*)&VT[d][2 * k2] = pack2(va, vb2);
      }
    }
    __syncthreads();

    // ---- S^T for two 16-key tiles (K and Q frags from LDS) ----
    v8f S0 = {0,0,0,0,0,0,0,0};
    v8f S1 = {0,0,0,0,0,0,0,0};
    const __bf16* k0p = &KT[r][0];
    const __bf16* k1p = &KT[16 + r][0];
#pragma unroll
    for (int dc = 0; dc < HD / 32; ++dc) {
      int off = dc * 32 + 8 * g;
      v16bf qfc = lds_frag(qp, off);
      v16bf ka0 = lds_frag(k0p, off);
      v16bf ka1 = lds_frag(k1p, off);
      S0 = __builtin_amdgcn_wmma_f32_16x16x32_bf16(false, ka0, false, qfc, (short)0, S0, false, false);
      S1 = __builtin_amdgcn_wmma_f32_16x16x32_bf16(false, ka1, false, qfc, (short)0, S1, false, false);
    }

    // ---- scale + mask bias; online softmax (per lane == per query) ----
    float smax = -1e30f;
#pragma unroll
    for (int t2 = 0; t2 < 2; ++t2) {
      v8f& S = t2 ? S1 : S0;
#pragma unroll
      for (int j = 0; j < 8; ++j) {
        int key = kb0 + j + 8 * g + 16 * t2;   // C layout: row = j + 8g
        bool valid = (key < PFX) ? (msk[b * PFX + key] != 0) : (key < TOTK);
        float s = S[j] * ATT_SCALE + (valid ? 0.f : NEGV);
        S[j] = s;
        smax = fmaxf(smax, s);
      }
    }
    smax = fmaxf(smax, __shfl_xor(smax, 16, 32));
    float mnew = fmaxf(mrun, smax);
    float corr = __expf(mrun - mnew);
    float psum = 0.f;
#pragma unroll
    for (int t2 = 0; t2 < 2; ++t2) {
      v8f& S = t2 ? S1 : S0;
#pragma unroll
      for (int j = 0; j < 8; ++j) { float p = __expf(S[j] - mnew); S[j] = p; psum += p; }
    }
    psum += __shfl_xor(psum, 16, 32);
    lrun = lrun * corr + psum;
    mrun = mnew;
#pragma unroll
    for (int dt = 0; dt < 16; ++dt)
#pragma unroll
      for (int j = 0; j < 8; ++j) Ot[dt][j] *= corr;

    // P^T (C-layout, 2 tiles) -> B-frag in-lane: e<8 from tile0, e>=8 tile1
    v16bf pb;
#pragma unroll
    for (int e = 0; e < 4; ++e) {
      v2bf lo = pack2(S0[2 * e], S0[2 * e + 1]);
      v2bf hi = pack2(S1[2 * e], S1[2 * e + 1]);
      pb[2 * e] = lo[0]; pb[2 * e + 1] = lo[1];
      pb[8 + 2 * e] = hi[0]; pb[9 + 2 * e] = hi[1];
    }

    // ---- O^T += V^T @ P^T over 16 d-tiles ----
#pragma unroll
    for (int dt = 0; dt < 16; ++dt) {
      v16bf vf = lds_frag(&VT[dt * 16 + r][0], 8 * g);
      Ot[dt] = __builtin_amdgcn_wmma_f32_16x16x32_bf16(false, vf, false, pb, (short)0, Ot[dt], false, false);
    }
    __syncthreads();
  }

  float inv = 1.0f / fmaxf(lrun, 1e-20f);
  if (qi < AH) {
    float* orow = outb + ((long long)(b * AH + qi)) * (NH * HD) + h * HD;
#pragma unroll
    for (int dt = 0; dt < 16; ++dt) {
      // O^T: element j -> d = dt*16 + 8g + j (contiguous 8 floats)
      v4fv s0 = {Ot[dt][0] * inv, Ot[dt][1] * inv, Ot[dt][2] * inv, Ot[dt][3] * inv};
      v4fv s1 = {Ot[dt][4] * inv, Ot[dt][5] * inv, Ot[dt][6] * inv, Ot[dt][7] * inv};
      *(v4fv*)(orow + dt * 16 + 8 * g)     = s0;
      *(v4fv*)(orow + dt * 16 + 8 * g + 4) = s1;
    }
  }
}

// =====================================================================
// small kernels
// =====================================================================
__global__ void offsets_kernel(const unsigned char* __restrict__ msk, int* __restrict__ offs)
{
  __shared__ int red[256];
  int b = blockIdx.x, tid = threadIdx.x;
  int s = 0;
  for (int j = tid; j < PFX; j += 256) s += (msk[b * PFX + j] != 0);
  red[tid] = s; __syncthreads();
  for (int st = 128; st > 0; st >>= 1) { if (tid < st) red[tid] += red[tid + st]; __syncthreads(); }
  if (tid == 0) offs[b] = red[0];
}

__global__ void timecond_kernel(const float* __restrict__ ts,
                                const float* __restrict__ Wti, const float* __restrict__ bti,
                                const float* __restrict__ Wto, const float* __restrict__ bto,
                                float* __restrict__ cond)
{
  __shared__ float tl[HID];
  __shared__ float h1[HID];
  int b = blockIdx.x, tid = threadIdx.x;
  float t = ts[b];
  for (int i = tid; i < HID / 2; i += 256) {
    float frac = (float)i / (float)(HID / 2 - 1);
    float period = 0.004f * __powf(1000.0f, frac);
    float x = (TWO_PI / period) * t;
    tl[i] = __sinf(x);
    tl[i + HID / 2] = __cosf(x);
  }
  __syncthreads();
  for (int j = tid; j < HID; j += 256) {
    float acc = bti[j];
    for (int k = 0; k < HID; ++k) acc += tl[k] * Wti[(long long)k * HID + j];
    h1[j] = acc / (1.f + __expf(-acc));            // silu
  }
  __syncthreads();
  for (int j = tid; j < HID; j += 256) {
    float acc = bto[j];
    for (int k = 0; k < HID; ++k) acc += h1[k] * Wto[(long long)k * HID + j];
    cond[(long long)b * HID + j] = acc / (1.f + __expf(-acc));
  }
}

__global__ void actin_kernel(const float* __restrict__ xt, const float* __restrict__ W,
                             const float* __restrict__ bias, float* __restrict__ hq)
{
  long long i = (long long)blockIdx.x * 256 + threadIdx.x;
  if (i >= (long long)NTOK * HID) return;
  int m = (int)(i / HID), j = (int)(i % HID);
  float acc = bias[j];
  const float* xr = xt + (long long)m * AD;
#pragma unroll
  for (int k = 0; k < AD; ++k) acc += xr[k] * W[k * HID + j];
  hq[i] = acc;
}

__global__ void adanorm_kernel(const float* __restrict__ hq, const float* __restrict__ ada,
                               float* __restrict__ outq)
{
  __shared__ float red[256];
  int m = blockIdx.x, tid = threadIdx.x;
  int b = m / AH;
  const float* row = hq + (long long)m * HID;
  float ss = 0.f;
  for (int j = tid; j < HID; j += 256) { float x = row[j]; ss += x * x; }
  red[tid] = ss; __syncthreads();
  for (int st = 128; st > 0; st >>= 1) { if (tid < st) red[tid] += red[tid + st]; __syncthreads(); }
  float rms = rsqrtf(red[0] / (float)HID + 1e-6f);
  const float* ar = ada + (long long)b * (3 * HID);
  for (int j = tid; j < HID; j += 256)
    outq[(long long)m * HID + j] = row[j] * rms * (1.f + ar[j]) + ar[HID + j];
}

__global__ void rope_kernel(float* __restrict__ buf, const int* __restrict__ offs, int heads)
{
  int i = blockIdx.x * 256 + threadIdx.x;
  int total = NTOK * heads * 128;
  if (i >= total) return;
  int m = i / (heads * 128);
  int rst = i % (heads * 128);
  int hh = rst / 128, d = rst % 128;
  int b = m / AH, t = m % AH;
  float pos = (float)(offs[b] + t);
  float inv = __powf(10000.f, -(float)d / 128.f);
  float th = pos * inv;
  float s = __sinf(th), c = __cosf(th);
  float* p = buf + (long long)m * heads * HD + hh * HD;
  float x1 = p[d], x2 = p[d + 128];
  p[d] = x1 * c - x2 * s;
  p[d + 128] = x2 * c + x1 * s;
}

__global__ void gatedres_kernel(float* __restrict__ hq, const float* __restrict__ xq,
                                const float* __restrict__ ada)
{
  long long i = (long long)blockIdx.x * 256 + threadIdx.x;
  if (i >= (long long)NTOK * HID) return;
  int m = (int)(i / HID), j = (int)(i % HID);
  int b = m / AH;
  hq[i] += ada[(long long)b * (3 * HID) + 2 * HID + j] * xq[i];
}

__global__ void gelumul_kernel(float* __restrict__ Gq, const float* __restrict__ Uq, long long n)
{
  long long i = (long long)blockIdx.x * 256 + threadIdx.x;
  if (i >= n) return;
  float x = Gq[i];
  float t = tanhf(0.7978845608f * (x + 0.044715f * x * x * x));
  Gq[i] = 0.5f * x * (1.f + t) * Uq[i];
}

__global__ void actout_kernel(const float* __restrict__ nq, const float* __restrict__ W,
                              const float* __restrict__ bias, float* __restrict__ outq)
{
  __shared__ float red[8][AD];
  int m = blockIdx.x, tid = threadIdx.x;
  int c = tid & 31, seg = tid >> 5;
  const float* row = nq + (long long)m * HID;
  float acc = 0.f;
  for (int k = seg * 128; k < seg * 128 + 128; ++k) acc += row[k] * W[k * AD + c];
  red[seg][c] = acc; __syncthreads();
  if (seg == 0) {
    float s = bias[c];
#pragma unroll
    for (int q = 0; q < 8; ++q) s += red[q][c];
    outq[(long long)m * AD + c] = s;
  }
}

// =====================================================================
extern "C" void kernel_launch(void* const* d_in, const int* in_sizes, int n_in,
                              void* d_out, int out_size, void* d_ws, size_t ws_size,
                              hipStream_t stream)
{
  const float* pk  = (const float*)d_in[0];
  const float* pv  = (const float*)d_in[1];
  const unsigned char* msk = (const unsigned char*)d_in[2];
  const float* xt  = (const float*)d_in[3];
  const float* ts  = (const float*)d_in[4];
  const float* Wq  = (const float*)d_in[5];
  const float* Wk  = (const float*)d_in[6];
  const float* Wv  = (const float*)d_in[7];
  const float* Wo  = (const float*)d_in[8];
  const float* Wg  = (const float*)d_in[9];
  const float* Wu  = (const float*)d_in[10];
  const float* Wd  = (const float*)d_in[11];
  const float* A1  = (const float*)d_in[12];
  const float* A2  = (const float*)d_in[13];
  const float* Af  = (const float*)d_in[14];
  const float* Wai = (const float*)d_in[15];
  const float* bai = (const float*)d_in[16];
  const float* Wao = (const float*)d_in[17];
  const float* bao = (const float*)d_in[18];
  const float* Wti = (const float*)d_in[19];
  const float* bti = (const float*)d_in[20];
  const float* Wto = (const float*)d_in[21];
  const float* bto = (const float*)d_in[22];

  float* ws = (float*)d_ws;
  size_t o = 0;
  auto alloc = [&](size_t n) { float* p = ws + o; o += n; return p; };
  float* cond = alloc((size_t)BATCH * HID);
  float* ada1 = alloc((size_t)NLAY * BATCH * 3 * HID);
  float* ada2 = alloc((size_t)NLAY * BATCH * 3 * HID);
  float* adaF = alloc((size_t)BATCH * 3 * HID);
  float* hbuf = alloc((size_t)NTOK * HID);
  float* nbuf = alloc((size_t)NTOK * HID);
  float* qbuf = alloc((size_t)NTOK * NH * HD);
  float* kbuf = alloc((size_t)NTOK * HD);
  float* vbuf = alloc((size_t)NTOK * HD);
  float* abuf = alloc((size_t)NTOK * NH * HD);
  float* obuf = alloc((size_t)NTOK * HID);
  float* Gbuf = alloc((size_t)NTOK * MLP);
  float* Ubuf = alloc((size_t)NTOK * MLP);
  int*   offs = (int*)(ws + o); o += 8;
  (void)in_sizes; (void)n_in; (void)out_size; (void)ws_size;

  const int MB = (NTOK + 31) / 32;   // 13 M-tiles for the 400-token GEMMs

  offsets_kernel<<<BATCH, 256, 0, stream>>>(msk, offs);
  timecond_kernel<<<BATCH, 256, 0, stream>>>(ts, Wti, bti, Wto, bto, cond);

  // AdaRMS condition projections (batched over layers in grid.z)
  {
    dim3 gz(1, (3 * HID) / 128, NLAY);
    gemm_kernel<HID, 3 * HID><<<gz, 128, 0, stream>>>(cond, A1, ada1, BATCH,
                                        (long long)HID * 3 * HID, (long long)BATCH * 3 * HID);
    gemm_kernel<HID, 3 * HID><<<gz, 128, 0, stream>>>(cond, A2, ada2, BATCH,
                                        (long long)HID * 3 * HID, (long long)BATCH * 3 * HID);
    dim3 gf(1, (3 * HID) / 128, 1);
    gemm_kernel<HID, 3 * HID><<<gf, 128, 0, stream>>>(cond, Af, adaF, BATCH, 0, 0);
  }
  actin_kernel<<<((size_t)NTOK * HID + 255) / 256, 256, 0, stream>>>(xt, Wai, bai, hbuf);

  const long long EW = (long long)NTOK * HID;
  for (int l = 0; l < NLAY; ++l) {
    const float* a1l = ada1 + (size_t)l * BATCH * 3 * HID;
    const float* a2l = ada2 + (size_t)l * BATCH * 3 * HID;

    adanorm_kernel<<<NTOK, 256, 0, stream>>>(hbuf, a1l, nbuf);
    { dim3 g(MB, (NH * HD) / 128, 1);
      gemm_kernel<HID, NH * HD><<<g, 128, 0, stream>>>(nbuf, Wq + (size_t)l * HID * NH * HD, qbuf, NTOK, 0, 0); }
    { dim3 g(MB, HD / 128, 1);
      gemm_kernel<HID, HD><<<g, 128, 0, stream>>>(nbuf, Wk + (size_t)l * HID * HD, kbuf, NTOK, 0, 0);
      gemm_kernel<HID, HD><<<g, 128, 0, stream>>>(nbuf, Wv + (size_t)l * HID * HD, vbuf, NTOK, 0, 0); }
    rope_kernel<<<(NTOK * NH * 128 + 255) / 256, 256, 0, stream>>>(qbuf, offs, NH);
    rope_kernel<<<(NTOK * 1 * 128 + 255) / 256, 256, 0, stream>>>(kbuf, offs, 1);
    { dim3 g(NH, BATCH, 1);
      attn_kernel<<<g, 128, 0, stream>>>(qbuf, kbuf, vbuf, pk, pv, msk, abuf, l); }
    { dim3 g(MB, HID / 128, 1);
      gemm_kernel<NH * HD, HID><<<g, 128, 0, stream>>>(abuf, Wo + (size_t)l * NH * HD * HID, obuf, NTOK, 0, 0); }
    gatedres_kernel<<<(EW + 255) / 256, 256, 0, stream>>>(hbuf, obuf, a1l);

    adanorm_kernel<<<NTOK, 256, 0, stream>>>(hbuf, a2l, nbuf);
    { dim3 g(MB, MLP / 128, 1);
      gemm_kernel<HID, MLP><<<g, 128, 0, stream>>>(nbuf, Wg + (size_t)l * HID * MLP, Gbuf, NTOK, 0, 0);
      gemm_kernel<HID, MLP><<<g, 128, 0, stream>>>(nbuf, Wu + (size_t)l * HID * MLP, Ubuf, NTOK, 0, 0); }
    gelumul_kernel<<<((long long)NTOK * MLP + 255) / 256, 256, 0, stream>>>(Gbuf, Ubuf, (long long)NTOK * MLP);
    { dim3 g(MB, HID / 128, 1);
      gemm_kernel<MLP, HID><<<g, 128, 0, stream>>>(Gbuf, Wd + (size_t)l * MLP * HID, obuf, NTOK, 0, 0); }
    gatedres_kernel<<<(EW + 255) / 256, 256, 0, stream>>>(hbuf, obuf, a2l);
  }

  adanorm_kernel<<<NTOK, 256, 0, stream>>>(hbuf, adaF, nbuf);
  actout_kernel<<<NTOK, 256, 0, stream>>>(nbuf, Wao, bao, (float*)d_out);
}
